// GCN_66829691125867
// MI455X (gfx1250) — matrix operand order repaired and checked by reading, not compiled, not run on hardware
//
#include <hip/hip_runtime.h>
#include <hip/hip_bf16.h>

typedef __attribute__((ext_vector_type(16))) __bf16 v16bf;
typedef __attribute__((ext_vector_type(8)))  __bf16 v8bf;
typedef __attribute__((ext_vector_type(8)))  float  v8f;

__device__ __forceinline__ __bf16 f2bf(float f) {
    unsigned u = __float_as_uint(f);
    unsigned r = (u + 0x7FFFu + ((u >> 16) & 1u)) >> 16;
    unsigned short s = (unsigned short)r;
    __bf16 out;
    __builtin_memcpy(&out, &s, 2);
    return out;
}
__device__ __forceinline__ v16bf cvt16(float4 a0, float4 a1, float4 a2, float4 a3) {
    v16bf af;
    af[0]=f2bf(a0.x);  af[1]=f2bf(a0.y);  af[2]=f2bf(a0.z);  af[3]=f2bf(a0.w);
    af[4]=f2bf(a1.x);  af[5]=f2bf(a1.y);  af[6]=f2bf(a1.z);  af[7]=f2bf(a1.w);
    af[8]=f2bf(a2.x);  af[9]=f2bf(a2.y);  af[10]=f2bf(a2.z); af[11]=f2bf(a2.w);
    af[12]=f2bf(a3.x); af[13]=f2bf(a3.y); af[14]=f2bf(a3.z); af[15]=f2bf(a3.w);
    return af;
}

// ---- CDNA5 async global->LDS copy (ASYNCcnt-tracked), 16 bytes per lane ----
__device__ __forceinline__ void async_copy16(unsigned lds_byte_off, const void* gptr) {
    asm volatile("global_load_async_to_lds_b128 %0, %1, off"
                 :: "v"(lds_byte_off), "v"(gptr) : "memory");
}
__device__ __forceinline__ void wait_async0() {
    asm volatile("s_wait_asynccnt 0x0" ::: "memory");
}
__device__ __forceinline__ unsigned lds_off_of(const void* p) {
    return (unsigned)(unsigned long long)p;   // flat LDS addr low 32 bits = LDS offset
}

// ---------------- weight prep ----------------
__global__ void k_prep_w1(const float* __restrict__ W1, __bf16* __restrict__ W1t) {
    int idx = blockIdx.x * blockDim.x + threadIdx.x;   // 512*128
    if (idx >= 512 * 128) return;
    int k = idx >> 7, n = idx & 127;
    W1t[(size_t)n * 512 + k] = f2bf(W1[(size_t)k * 128 + n]);
}
__global__ void k_prep_w2(const float* __restrict__ W2, __bf16* __restrict__ W2t) {
    int idx = blockIdx.x * blockDim.x + threadIdx.x;   // 48*128
    if (idx >= 48 * 128) return;
    int n = idx >> 7, k = idx & 127;
    float v = (n < 40) ? W2[(size_t)k * 40 + n] : 0.0f;
    W2t[(size_t)n * 128 + k] = f2bf(v);
}

// ---------------- degree / norm ----------------
__global__ void k_deg_init(float* __restrict__ deg, int N) {
    int i = blockIdx.x * blockDim.x + threadIdx.x;
    if (i < N) deg[i] = 1.0f;
}
__global__ void k_deg_acc(const int* __restrict__ ei, const float* __restrict__ w,
                          float* __restrict__ deg, int E) {
    int e = blockIdx.x * blockDim.x + threadIdx.x;
    if (e < E) unsafeAtomicAdd(&deg[ei[E + e]], w[e]);
}
__global__ void k_dinv(float* __restrict__ deg, int N) {
    int i = blockIdx.x * blockDim.x + threadIdx.x;
    if (i < N) deg[i] = rsqrtf(deg[i]);
}
__global__ void k_norm(const int* __restrict__ ei, const float* __restrict__ w,
                       const float* __restrict__ dinv, float* __restrict__ norm, int E) {
    int e = blockIdx.x * blockDim.x + threadIdx.x;
    if (e < E) norm[e] = dinv[ei[e]] * w[e] * dinv[ei[E + e]];
}

// ---------------- GEMM1: h1[N][128] = X[N][512] @ W1t^T ----------------
// 64-wide K slabs of B async-staged to LDS (double buffered); A 2-deep pipelined;
// B fragments preloaded so the 8 WMMAs issue back-to-back.
#define SB1_STRIDE 72   // 64 elems + 8 pad; 36*n mod 64 distinct for n=0..15 => conflict-free
__global__ __launch_bounds__(256) void k_gemm1(const float* __restrict__ X,
                                               const __bf16* __restrict__ Wt,
                                               float* __restrict__ H1, int N) {
    __shared__ __bf16 sB[2][128 * SB1_STRIDE];

    const int lane = threadIdx.x & 31;
    const int wave = threadIdx.x >> 5;
    const int m0   = blockIdx.x * 128 + wave * 16;
    const int rlo  = lane & 15;
    const int hi   = lane >> 4;
    const int kOffA = hi * 8;

    int arow = m0 + rlo; if (arow >= N) arow = N - 1;
    const float* ap = X + (size_t)arow * 512;

    // stage W1t[:, kb..kb+63] into sB[b] : 128 rows x 128B = 1024 x 16B chunks
    auto stage = [&](int kb, int b) {
        #pragma unroll
        for (int i = 0; i < 4; ++i) {
            int c = (int)threadIdx.x + i * 256;    // 0..1023
            int row = c >> 3, q = c & 7;
            unsigned lds = lds_off_of(&sB[b][row * SB1_STRIDE + q * 8]);
            async_copy16(lds, Wt + (size_t)row * 512 + kb + q * 8);
        }
    };

    v8f acc[8];
    #pragma unroll
    for (int t = 0; t < 8; ++t)
        #pragma unroll
        for (int j = 0; j < 8; ++j) acc[t][j] = 0.0f;

    // one 32-wide sub-step: preload 8 B fragments, then 8 back-to-back WMMAs
    int buf = 0;
    auto substep = [&](int ks, v16bf af) {
        v16bf bfr[8];
        #pragma unroll
        for (int nt = 0; nt < 8; ++nt) {
            const __bf16* bp = &sB[buf][(nt * 16 + rlo) * SB1_STRIDE + ks + hi * 8];
            v8bf b0 = *(const v8bf*)bp;
            v8bf b1 = *(const v8bf*)(bp + 16);
            #pragma unroll
            for (int j = 0; j < 8; ++j) { bfr[nt][j] = b0[j]; bfr[nt][j + 8] = b1[j]; }
        }
        #pragma unroll
        for (int nt = 0; nt < 8; ++nt)
            acc[nt] = __builtin_amdgcn_wmma_f32_16x16x32_bf16(
                false, af, false, bfr[nt], (short)0, acc[nt], false, false);
    };

    // prologue: A fragments for k=0 and k=32, async-stage slab kb=0
    float4 c0 = *(const float4*)(ap + kOffA);
    float4 c1 = *(const float4*)(ap + kOffA + 4);
    float4 c2 = *(const float4*)(ap + kOffA + 16);
    float4 c3 = *(const float4*)(ap + kOffA + 20);
    float4 n0 = *(const float4*)(ap + 32 + kOffA);
    float4 n1 = *(const float4*)(ap + 32 + kOffA + 4);
    float4 n2 = *(const float4*)(ap + 32 + kOffA + 16);
    float4 n3 = *(const float4*)(ap + 32 + kOffA + 20);
    stage(0, 0);
    wait_async0();
    __syncthreads();

    for (int kb = 0; kb < 512; kb += 64) {
        if (kb + 64 < 512) stage(kb + 64, buf ^ 1);

        // sub-step 0: k = kb
        v16bf af0 = cvt16(c0, c1, c2, c3);
        int kn0 = (kb + 64 < 512) ? (kb + 64) : 0;
        c0 = *(const float4*)(ap + kn0 + kOffA);
        c1 = *(const float4*)(ap + kn0 + kOffA + 4);
        c2 = *(const float4*)(ap + kn0 + kOffA + 16);
        c3 = *(const float4*)(ap + kn0 + kOffA + 20);
        substep(0, af0);

        // sub-step 1: k = kb + 32
        v16bf af1 = cvt16(n0, n1, n2, n3);
        int kn1 = (kb + 96 < 512) ? (kb + 96) : 0;
        n0 = *(const float4*)(ap + kn1 + kOffA);
        n1 = *(const float4*)(ap + kn1 + kOffA + 4);
        n2 = *(const float4*)(ap + kn1 + kOffA + 16);
        n3 = *(const float4*)(ap + kn1 + kOffA + 20);
        substep(32, af1);

        wait_async0();
        __syncthreads();
        buf ^= 1;
    }

    if (m0 + 16 <= N) {
        #pragma unroll
        for (int nt = 0; nt < 8; ++nt)
            #pragma unroll
            for (int r = 0; r < 8; ++r)
                H1[(size_t)(m0 + hi * 8 + r) * 128 + nt * 16 + rlo] = acc[nt][r];
    } else {
        #pragma unroll
        for (int nt = 0; nt < 8; ++nt)
            #pragma unroll
            for (int r = 0; r < 8; ++r) {
                int row = m0 + hi * 8 + r;
                if (row < N) H1[(size_t)row * 128 + nt * 16 + rlo] = acc[nt][r];
            }
    }
}

// ---------------- aggregation ----------------
__global__ void k_agg_init(const float* __restrict__ h, const float* __restrict__ dinv,
                           const float* __restrict__ bias, float* __restrict__ out,
                           int N, int F, int ldh, int ldo) {
    size_t tid = (size_t)blockIdx.x * blockDim.x + threadIdx.x;
    if (tid >= (size_t)N * F) return;
    int i = (int)(tid / F), f = (int)(tid % F);
    float d = dinv[i];
    out[(size_t)i * ldo + f] = d * d * h[(size_t)i * ldh + f] + bias[f];
}
__global__ void k_agg_edges128(const int* __restrict__ ei, const float* __restrict__ norm,
                               const float* __restrict__ h, float* __restrict__ out, int E) {
    size_t tid = (size_t)blockIdx.x * blockDim.x + threadIdx.x;
    if (tid >= ((size_t)E << 7)) return;
    int e = (int)(tid >> 7), f = (int)(tid & 127);
    int row = ei[e], col = ei[E + e];
    unsafeAtomicAdd(&out[((size_t)col << 7) + f], norm[e] * h[((size_t)row << 7) + f]);
}
__global__ void k_agg_edges(const int* __restrict__ ei, const float* __restrict__ norm,
                            const float* __restrict__ h, float* __restrict__ out,
                            int E, int F, int ldh, int ldo) {
    size_t tid = (size_t)blockIdx.x * blockDim.x + threadIdx.x;
    if (tid >= (size_t)E * F) return;
    int e = (int)(tid / F), f = (int)(tid % F);
    int row = ei[e], col = ei[E + e];
    unsafeAtomicAdd(&out[(size_t)col * ldo + f], norm[e] * h[(size_t)row * ldh + f]);
}

__global__ void k_relu_bf16(const float* __restrict__ in, __bf16* __restrict__ out, size_t n) {
    size_t i = (size_t)blockIdx.x * blockDim.x + threadIdx.x;
    if (i < n) out[i] = f2bf(fmaxf(in[i], 0.0f));
}

// ---------------- GEMM2: h2[N][48] = X1b[N][128] @ W2t^T ----------------
#define SB2_STRIDE 136  // 128 + 8 pad; 4*n mod 64 distinct => conflict-free
__global__ __launch_bounds__(256) void k_gemm2(const __bf16* __restrict__ X,
                                               const __bf16* __restrict__ Wt,
                                               float* __restrict__ H2, int N) {
    __shared__ __bf16 sB[48 * SB2_STRIDE];

    const int lane = threadIdx.x & 31;
    const int wave = threadIdx.x >> 5;
    const int m0   = blockIdx.x * 128 + wave * 16;
    const int rlo  = lane & 15;
    const int hi   = lane >> 4;
    const int kOffA = hi * 8;

    // stage entire W2t (48x128) once: 768 x 16B chunks
    #pragma unroll
    for (int i = 0; i < 3; ++i) {
        int c = (int)threadIdx.x + i * 256;   // 0..767
        int row = c >> 4, q = c & 15;
        unsigned lds = lds_off_of(&sB[row * SB2_STRIDE + q * 8]);
        async_copy16(lds, Wt + (size_t)row * 128 + q * 8);
    }

    int arow = m0 + rlo; if (arow >= N) arow = N - 1;
    const __bf16* ap = X + (size_t)arow * 128;

    v8bf al[4][2];
    #pragma unroll
    for (int s = 0; s < 4; ++s) {
        al[s][0] = *(const v8bf*)(ap + s * 32 + kOffA);
        al[s][1] = *(const v8bf*)(ap + s * 32 + kOffA + 16);
    }

    v8f acc[3];
    #pragma unroll
    for (int t = 0; t < 3; ++t)
        #pragma unroll
        for (int j = 0; j < 8; ++j) acc[t][j] = 0.0f;

    wait_async0();
    __syncthreads();

    #pragma unroll
    for (int s = 0; s < 4; ++s) {
        v16bf af;
        #pragma unroll
        for (int j = 0; j < 8; ++j) { af[j] = al[s][0][j]; af[j + 8] = al[s][1][j]; }
        v16bf bfr[3];
        #pragma unroll
        for (int nt = 0; nt < 3; ++nt) {
            const __bf16* bp = &sB[(nt * 16 + rlo) * SB2_STRIDE + s * 32 + hi * 8];
            v8bf b0 = *(const v8bf*)bp;
            v8bf b1 = *(const v8bf*)(bp + 16);
            #pragma unroll
            for (int j = 0; j < 8; ++j) { bfr[nt][j] = b0[j]; bfr[nt][j + 8] = b1[j]; }
        }
        #pragma unroll
        for (int nt = 0; nt < 3; ++nt)
            acc[nt] = __builtin_amdgcn_wmma_f32_16x16x32_bf16(
                false, af, false, bfr[nt], (short)0, acc[nt], false, false);
    }

    if (m0 + 16 <= N) {
        #pragma unroll
        for (int nt = 0; nt < 3; ++nt)
            #pragma unroll
            for (int r = 0; r < 8; ++r)
                H2[(size_t)(m0 + hi * 8 + r) * 48 + nt * 16 + rlo] = acc[nt][r];
    } else {
        #pragma unroll
        for (int nt = 0; nt < 3; ++nt)
            #pragma unroll
            for (int r = 0; r < 8; ++r) {
                int row = m0 + hi * 8 + r;
                if (row < N) H2[(size_t)row * 48 + nt * 16 + rlo] = acc[nt][r];
            }
    }
}

// ---------------- log_softmax over 40 classes, one wave per node ----------------
__global__ __launch_bounds__(256) void k_logsoftmax(const float* __restrict__ in,
                                                    float* __restrict__ out, int N) {
    int lane = threadIdx.x & 31;
    int node = blockIdx.x * 8 + (threadIdx.x >> 5);
    if (node >= N) return;
    const float* p = in + (size_t)node * 40;
    float v0 = p[lane];
    float v1 = (lane < 8) ? p[lane + 32] : -__builtin_inff();
    float m = fmaxf(v0, v1);
    #pragma unroll
    for (int off = 16; off > 0; off >>= 1) m = fmaxf(m, __shfl_xor(m, off, 32));
    float s = __expf(v0 - m) + ((lane < 8) ? __expf(v1 - m) : 0.0f);
    #pragma unroll
    for (int off = 16; off > 0; off >>= 1) s += __shfl_xor(s, off, 32);
    float l = __logf(s);
    out[(size_t)node * 40 + lane] = v0 - m - l;
    if (lane < 8) out[(size_t)node * 40 + 32 + lane] = v1 - m - l;
}

// ---------------- host launcher ----------------
static inline int cdiv(long long a, long long b) { return (int)((a + b - 1) / b); }

extern "C" void kernel_launch(void* const* d_in, const int* in_sizes, int n_in,
                              void* d_out, int out_size, void* d_ws, size_t ws_size,
                              hipStream_t stream) {
    const float* X   = (const float*)d_in[0];
    const int*   ei  = (const int*)d_in[1];
    const float* ew  = (const float*)d_in[2];
    const float* W1  = (const float*)d_in[3];
    const float* b1  = (const float*)d_in[4];
    const float* W2  = (const float*)d_in[5];
    const float* b2  = (const float*)d_in[6];
    float* OUT = (float*)d_out;

    const int N = in_sizes[0] / 512;
    const int E = in_sizes[2];

    char* p = (char*)d_ws;
    auto carve = [&](size_t bytes) { char* r = p; p += (bytes + 255) & ~(size_t)255; return r; };
    __bf16* W1t  = (__bf16*)carve((size_t)128 * 512 * 2);
    __bf16* W2t  = (__bf16*)carve((size_t)48 * 128 * 2);
    float*  dinv = (float*)carve((size_t)N * 4);
    float*  nrm  = (float*)carve((size_t)E * 4);
    float*  h1   = (float*)carve((size_t)N * 128 * 4);
    float*  o1   = (float*)carve((size_t)N * 128 * 4);
    __bf16* x1b  = (__bf16*)carve((size_t)N * 128 * 2);
    float*  h2   = (float*)carve((size_t)N * 48 * 4);
    float*  o2   = (float*)carve((size_t)N * 40 * 4);

    const int B = 256;
    k_prep_w1<<<cdiv(512 * 128, B), B, 0, stream>>>(W1, W1t);
    k_prep_w2<<<cdiv(48 * 128, B), B, 0, stream>>>(W2, W2t);

    k_deg_init<<<cdiv(N, B), B, 0, stream>>>(dinv, N);
    k_deg_acc<<<cdiv(E, B), B, 0, stream>>>(ei, ew, dinv, E);
    k_dinv<<<cdiv(N, B), B, 0, stream>>>(dinv, N);
    k_norm<<<cdiv(E, B), B, 0, stream>>>(ei, ew, dinv, nrm, E);

    k_gemm1<<<cdiv(N, 128), B, 0, stream>>>(X, W1t, h1, N);
    k_agg_init<<<cdiv((long long)N * 128, B), B, 0, stream>>>(h1, dinv, b1, o1, N, 128, 128, 128);
    k_agg_edges128<<<cdiv((long long)E * 128, B), B, 0, stream>>>(ei, nrm, h1, o1, E);
    k_relu_bf16<<<cdiv((long long)N * 128, B), B, 0, stream>>>(o1, x1b, (size_t)N * 128);

    k_gemm2<<<cdiv(N, 128), B, 0, stream>>>(x1b, W2t, h2, N);
    k_agg_init<<<cdiv((long long)N * 40, B), B, 0, stream>>>(h2, dinv, b2, o2, N, 40, 48, 40);
    k_agg_edges<<<cdiv((long long)E * 40, B), B, 0, stream>>>(ei, nrm, h2, o2, E, 40, 48, 40);

    k_logsoftmax<<<cdiv(N, 8), B, 0, stream>>>(o2, OUT, N);
}
